// MultiHeadAttention_40149354283868
// MI455X (gfx1250) — matrix operand order repaired and checked by reading, not compiled, not run on hardware
//
#include <hip/hip_runtime.h>
#include <math.h>

// ---------------------------------------------------------------------------
// MHA for MI455X (gfx1250): bf16 WMMA everywhere, f32 accumulate.
//   pass 1: cast activations f32->bf16, cast+transpose weights f32->bf16
//   pass 2: Q/K/V projections   (bf16 GEMM, 128x256 block tile, K-step 32)
//   pass 3: fused flash-attention per (b,h), 64 q-rows per block
//   pass 4: output projection + bias -> f32 d_out
// Data movement: GLOBAL_LOAD_ASYNC_TO_LDS_B128 (ASYNCcnt-tracked, bypasses
// VGPRs) with double-buffered LDS ping-pong: next tile's async loads are in
// flight while the current tile's WMMAs execute.
// ---------------------------------------------------------------------------

#define D_MODEL 2048
#define N_HEADS 16
#define D_K     128
#define BATCH   2
#define SEQ     2048
#define M_TOT   (BATCH * SEQ)   // 4096

typedef __attribute__((ext_vector_type(16))) __bf16 v16bf;
typedef __attribute__((ext_vector_type(8)))  float  v8f;
typedef __attribute__((ext_vector_type(4)))  int    v4i;

union Frag  { uint4 u[2]; v16bf v; };     // 32B: one WMMA bf16 operand
union Pack8 { uint4 u;    __bf16 e[8]; }; // 16B: 8 bf16

__device__ __forceinline__ v8f wmma_bf16(v16bf a, v16bf b, v8f c) {
  // D = A(16x32 bf16) * B(32x16 bf16) + C(16x16 f32)
  return __builtin_amdgcn_wmma_f32_16x16x32_bf16(
      /*neg_a=*/false, a, /*neg_b=*/false, b,
      /*c_mod=*/(short)0, c, /*reuse_a=*/false, /*reuse_b=*/false);
}

// ---- async global->LDS copy (gfx1250 GLOBAL_LOAD_ASYNC_TO_LDS_B128) -------
#if defined(__has_builtin)
#if __has_builtin(__builtin_amdgcn_global_load_async_to_lds_b128)
#define HAVE_ASYNC_LDS 1
#endif
#endif

#ifdef HAVE_ASYNC_LDS
typedef __attribute__((address_space(1))) v4i* gv4i_p;  // global 128-bit
typedef __attribute__((address_space(3))) v4i* lv4i_p;  // LDS 128-bit
#endif

__device__ __forceinline__ void cp16(const void* g, void* l) {
#ifdef HAVE_ASYNC_LDS
  __builtin_amdgcn_global_load_async_to_lds_b128((gv4i_p)g, (lv4i_p)l,
                                                 /*offset=*/0, /*cpol=*/0);
#else
  *(uint4*)l = *(const uint4*)g;
#endif
}
__device__ __forceinline__ void cp_wait() {
#ifdef HAVE_ASYNC_LDS
#if __has_builtin(__builtin_amdgcn_s_wait_asynccnt)
  __builtin_amdgcn_s_wait_asynccnt(0);
#else
  asm volatile("s_wait_asynccnt 0x0" ::: "memory");
#endif
#endif
}

// A-fragment K order used here (16-bit A 16x32, contiguous, not the ISA table order):
//   elems j -> K = half*16 + j (valid because B uses the same order)
// B-fragment (same order as A): elems j -> K = half*16 + j.
// Both load as two contiguous uint4 when K is the contiguous LDS dimension.

// ---------------------------------------------------------------------------
__global__ void cast_f32_bf16(const float* __restrict__ src,
                              __bf16* __restrict__ dst, int n) {
  int i = blockIdx.x * blockDim.x + threadIdx.x;
  int stride = gridDim.x * blockDim.x;
  for (; i < n; i += stride) dst[i] = (__bf16)src[i];
}

// W:[K][N] f32 row-major  ->  Wt:[N][K] bf16  (K = N = D_MODEL)
__global__ void transpose_cast(const float* __restrict__ W,
                               __bf16* __restrict__ Wt) {
  __shared__ float tile[32][33];
  const int k0 = blockIdx.y * 32, n0 = blockIdx.x * 32;
  const int tx = threadIdx.x & 31, ty = threadIdx.x >> 5;  // 256 thr: ty 0..7
  #pragma unroll
  for (int r = ty; r < 32; r += 8)
    tile[r][tx] = W[(size_t)(k0 + r) * D_MODEL + n0 + tx];
  __syncthreads();
  #pragma unroll
  for (int r = ty; r < 32; r += 8)
    Wt[(size_t)(n0 + r) * D_MODEL + k0 + tx] = (__bf16)tile[tx][r];
}

// ---------------------------------------------------------------------------
// C[M][N] = A[M][K] * Bt[N][K]^T + bias.
// 256 thr = 8 waves; block tile 128(M) x 256(N); wave tile 64x64 (4x4 accums);
// K-step 32, double-buffered LDS: tile kt+32 streams in (async) under the
// 16 WMMAs of tile kt.
template <bool OUT_BF16>
__global__ void __launch_bounds__(256, 1)
gemm_bf16(const __bf16* __restrict__ A,
          const __bf16* __restrict__ Bt,
          const float* __restrict__ bias,
          void* __restrict__ Cout,
          int M, int N, int K) {
  __shared__ __align__(16) __bf16 As[2][128 * 32];   // 16 KB
  __shared__ __align__(16) __bf16 Bs[2][256 * 32];   // 32 KB

  const int tid  = threadIdx.x;
  const int lane = tid & 31, w = tid >> 5;
  const int half = lane >> 4, lm = lane & 15;
  const int wr = w >> 2, wc = w & 3;              // 2 x 4 wave grid
  const int m0 = blockIdx.y * 128, n0 = blockIdx.x * 256;

  auto stage = [&](int sb, int kk) {
    #pragma unroll
    for (int i = 0; i < 2; ++i) {               // A tile: 512 16B chunks
      const int c = tid + i * 256;
      const int row = c >> 2, koff = (c & 3) << 3;
      cp16(&A[(size_t)(m0 + row) * K + kk + koff],
           &As[sb][row * 32 + koff]);
    }
    #pragma unroll
    for (int i = 0; i < 4; ++i) {               // B tile: 1024 16B chunks
      const int c = tid + i * 256;
      const int row = c >> 2, koff = (c & 3) << 3;
      cp16(&Bt[(size_t)(n0 + row) * K + kk + koff],
           &Bs[sb][row * 32 + koff]);
    }
  };

  const v8f vzero = {0.f, 0.f, 0.f, 0.f, 0.f, 0.f, 0.f, 0.f};
  v8f acc[4][4];
  #pragma unroll
  for (int i = 0; i < 4; ++i)
    #pragma unroll
    for (int j = 0; j < 4; ++j) acc[i][j] = vzero;

  stage(0, 0);
  cp_wait();
  __syncthreads();

  int buf = 0;
  for (int kt = 0; kt < K; kt += 32, buf ^= 1) {
    if (kt + 32 < K) stage(buf ^ 1, kt + 32);   // fire-and-forget next tile

    Frag bf[4];
    #pragma unroll
    for (int j = 0; j < 4; ++j) {
      const int n = wc * 64 + j * 16 + lm;
      bf[j].u[0] = *(const uint4*)&Bs[buf][n * 32 + half * 16];
      bf[j].u[1] = *(const uint4*)&Bs[buf][n * 32 + half * 16 + 8];
    }
    // double-buffered A fragments: overlap ds_load with v_wmma
    Frag af[2];
    {
      const int r = wr * 64 + lm;
      af[0].u[0] = *(const uint4*)&As[buf][r * 32 + half * 16];
      af[0].u[1] = *(const uint4*)&As[buf][r * 32 + half * 16 + 8];
    }
    #pragma unroll
    for (int i = 0; i < 4; ++i) {
      if (i < 3) {
        const int r = wr * 64 + (i + 1) * 16 + lm;
        af[(i + 1) & 1].u[0] = *(const uint4*)&As[buf][r * 32 + half * 16];
        af[(i + 1) & 1].u[1] =
            *(const uint4*)&As[buf][r * 32 + half * 16 + 8];
      }
      #pragma unroll
      for (int j = 0; j < 4; ++j)
        acc[i][j] = wmma_bf16(af[i & 1].v, bf[j].v, acc[i][j]);
    }

    cp_wait();         // next tile fully in LDS (this wave's asyncs)
    __syncthreads();   // ... and everyone else's; buf[kt] free for reuse
  }

  // C layout: elem r, half -> row = r + 8*half ; col = lm
  #pragma unroll
  for (int i = 0; i < 4; ++i) {
    const int mrow = m0 + wr * 64 + i * 16 + half * 8;
    #pragma unroll
    for (int j = 0; j < 4; ++j) {
      const int col = n0 + wc * 64 + j * 16 + lm;
      const float bv = bias ? bias[col] : 0.f;
      #pragma unroll
      for (int r = 0; r < 8; ++r) {
        const float v = acc[i][j][r] + bv;
        if (OUT_BF16)
          ((__bf16*)Cout)[(size_t)(mrow + r) * N + col] = (__bf16)v;
        else
          ((float*)Cout)[(size_t)(mrow + r) * N + col] = v;
      }
    }
  }
}

// ---------------------------------------------------------------------------
// Fused flash attention: block = 128 thr = 4 waves; block covers 64 q-rows of
// one (b,h); each wave owns 16 q-rows x 128 d. Online softmax in f32.
// K/V tiles double-buffered: next K streams via async-to-LDS, next V rides in
// VGPRs during compute and is scattered (transposed) afterwards.
__global__ void __launch_bounds__(128, 1)
attn_fused(const __bf16* __restrict__ Qp,
           const __bf16* __restrict__ Kp,
           const __bf16* __restrict__ Vp,
           __bf16* __restrict__ Ao) {
  __shared__ __align__(16) __bf16 Kt[2][32 * 128];  // [key][d]       16 KB
  __shared__ __align__(16) __bf16 Vt[2][128 * 32];  // [d][key] (T)   16 KB
  __shared__ __align__(16) __bf16 Pl[4][16 * 32];   // per-wave P      4 KB

  const int tid  = threadIdx.x;
  const int lane = tid & 31, w = tid >> 5;
  const int half = lane >> 4, lm = lane & 15;
  const int bh = blockIdx.y;
  const int b = bh >> 4, h = bh & 15;
  const int q0 = blockIdx.x * 64 + w * 16;

  // Q fragments: rows q0+lm, 4 frags cover d = 0..127 (K-step 32 each)
  const size_t rowQ = (size_t)(b * SEQ + q0 + lm) * D_MODEL + h * D_K;
  Frag qf[4];
  #pragma unroll
  for (int f = 0; f < 4; ++f) {
    qf[f].u[0] = *(const uint4*)&Qp[rowQ + f * 32 + half * 16];
    qf[f].u[1] = *(const uint4*)&Qp[rowQ + f * 32 + half * 16 + 8];
  }

  const v8f vzero = {0.f, 0.f, 0.f, 0.f, 0.f, 0.f, 0.f, 0.f};
  v8f o[8];
  float mi[8], li[8];
  #pragma unroll
  for (int c = 0; c < 8; ++c) o[c] = vzero;
  #pragma unroll
  for (int r = 0; r < 8; ++r) { mi[r] = -1e30f; li[r] = 0.f; }

  const float scale = 0.08838834764831845f;  // 1/sqrt(128)

  // prologue: stage key tile 0 into buffer 0
  #pragma unroll
  for (int i = 0; i < 4; ++i) {
    const int c = tid + i * 128;
    const int kr = c >> 4, doff = (c & 15) << 3;
    const size_t g = (size_t)(b * SEQ + kr) * D_MODEL + h * D_K + doff;
    cp16(&Kp[g], &Kt[0][kr * 128 + doff]);
    Pack8 vv; vv.u = *(const uint4*)&Vp[g];
    #pragma unroll
    for (int e = 0; e < 8; ++e) Vt[0][(doff + e) * 32 + kr] = vv.e[e];
  }
  cp_wait();
  __syncthreads();

  for (int t0 = 0; t0 < SEQ; t0 += 32) {
    const int buf = (t0 >> 5) & 1;
    const bool has_next = (t0 + 32 < SEQ);

    // fire next tile: K via async-to-LDS; V into registers (scattered later)
    Pack8 vv[4];
    if (has_next) {
      #pragma unroll
      for (int i = 0; i < 4; ++i) {
        const int c = tid + i * 128;
        const int kr = c >> 4, doff = (c & 15) << 3;
        const size_t g =
            (size_t)(b * SEQ + t0 + 32 + kr) * D_MODEL + h * D_K + doff;
        cp16(&Kp[g], &Kt[buf ^ 1][kr * 128 + doff]);
        vv[i].u = *(const uint4*)&Vp[g];
      }
    }

    // scores: two 16x16 tiles (keys t0+0..15, t0+16..31); K-frag pipeline
    v8f s[2]; s[0] = vzero; s[1] = vzero;
    Frag kf[2];
    kf[0].u[0] = *(const uint4*)&Kt[buf][lm * 128 + half * 16];
    kf[0].u[1] = *(const uint4*)&Kt[buf][lm * 128 + half * 16 + 8];
    #pragma unroll
    for (int u = 0; u < 8; ++u) {
      if (u < 7) {
        const int st = (u + 1) >> 2, f = (u + 1) & 3;
        const int n = st * 16 + lm;
        kf[(u + 1) & 1].u[0] =
            *(const uint4*)&Kt[buf][n * 128 + f * 32 + half * 16];
        kf[(u + 1) & 1].u[1] =
            *(const uint4*)&Kt[buf][n * 128 + f * 32 + half * 16 + 8];
      }
      s[u >> 2] = wmma_bf16(qf[u & 3].v, kf[u & 1].v, s[u >> 2]);
    }

    // online softmax; row r+8*half lives in one 16-lane group => width-16
    // shfl reductions. Store P to LDS immediately (short live ranges).
    #pragma unroll
    for (int r = 0; r < 8; ++r) {
      const float a = s[0][r] * scale, c = s[1][r] * scale;
      float mt = fmaxf(a, c);
      #pragma unroll
      for (int xm = 1; xm < 16; xm <<= 1)
        mt = fmaxf(mt, __shfl_xor(mt, xm, 32));
      const float mn = fmaxf(mi[r], mt);
      const float alpha = __expf(mi[r] - mn);
      const float p0 = __expf(a - mn);
      const float p1 = __expf(c - mn);
      Pl[w][(half * 8 + r) * 32 + lm]      = (__bf16)p0;
      Pl[w][(half * 8 + r) * 32 + 16 + lm] = (__bf16)p1;
      float rs = p0 + p1;
      #pragma unroll
      for (int xm = 1; xm < 16; xm <<= 1)
        rs += __shfl_xor(rs, xm, 32);
      li[r] = li[r] * alpha + rs;
      mi[r] = mn;
      #pragma unroll
      for (int c2 = 0; c2 < 8; ++c2) o[c2][r] *= alpha;
    }

    // Reload P as A fragment (same-wave DS ops are in-order: no barrier).
    Frag pf;
    pf.u[0] = *(const uint4*)&Pl[w][lm * 32 + half * 16];
    pf.u[1] = *(const uint4*)&Pl[w][lm * 32 + half * 16 + 8];

    // O += P(16x32) x V(32x128): V-frag pipeline over 8 column tiles
    Frag vf[2];
    vf[0].u[0] = *(const uint4*)&Vt[buf][lm * 32 + half * 16];
    vf[0].u[1] = *(const uint4*)&Vt[buf][lm * 32 + half * 16 + 8];
    #pragma unroll
    for (int ct = 0; ct < 8; ++ct) {
      if (ct < 7) {
        const int n = (ct + 1) * 16 + lm;
        vf[(ct + 1) & 1].u[0] = *(const uint4*)&Vt[buf][n * 32 + half * 16];
        vf[(ct + 1) & 1].u[1] =
            *(const uint4*)&Vt[buf][n * 32 + half * 16 + 8];
      }
      o[ct] = wmma_bf16(pf.v, vf[ct & 1].v, o[ct]);
    }

    // scatter next V tile (transposed) now that its loads have had the whole
    // compute phase to land
    if (has_next) {
      #pragma unroll
      for (int i = 0; i < 4; ++i) {
        const int c = tid + i * 128;
        const int kr = c >> 4, doff = (c & 15) << 3;
        #pragma unroll
        for (int e = 0; e < 8; ++e)
          Vt[buf ^ 1][(doff + e) * 32 + kr] = vv[i].e[e];
      }
    }
    cp_wait();
    __syncthreads();
  }

  // normalize and store bf16 [B*S][D_MODEL]
  #pragma unroll
  for (int r = 0; r < 8; ++r) {
    const float inv = 1.f / li[r];
    const size_t row =
        (size_t)(b * SEQ + q0 + half * 8 + r) * D_MODEL + h * D_K;
    #pragma unroll
    for (int ct = 0; ct < 8; ++ct)
      Ao[row + ct * 16 + lm] = (__bf16)(o[ct][r] * inv);
  }
}

// ---------------------------------------------------------------------------
extern "C" void kernel_launch(void* const* d_in, const int* in_sizes, int n_in,
                              void* d_out, int out_size, void* d_ws,
                              size_t ws_size, hipStream_t stream) {
  (void)in_sizes; (void)n_in; (void)out_size; (void)ws_size;
  const float* q  = (const float*)d_in[0];
  const float* k  = (const float*)d_in[1];
  const float* v  = (const float*)d_in[2];
  const float* Wq = (const float*)d_in[3];
  const float* bq = (const float*)d_in[4];
  const float* Wk = (const float*)d_in[5];
  const float* bk = (const float*)d_in[6];
  const float* Wv = (const float*)d_in[7];
  const float* bv = (const float*)d_in[8];
  const float* Wo = (const float*)d_in[9];
  const float* bo = (const float*)d_in[10];
  float* out = (float*)d_out;

  const size_t ACT = (size_t)M_TOT * D_MODEL;     // 8,388,608 elems
  const size_t WEL = (size_t)D_MODEL * D_MODEL;   // 4,194,304 elems
  char* p = (char*)d_ws;
  __bf16* Xq  = (__bf16*)p;                 p += ACT * 2;
  __bf16* Xk  = (__bf16*)p;                 p += ACT * 2;
  __bf16* Xv  = (__bf16*)p;                 p += ACT * 2;
  __bf16* WqT = (__bf16*)p;                 p += WEL * 2;
  __bf16* WkT = (__bf16*)p;                 p += WEL * 2;
  __bf16* WvT = (__bf16*)p;                 p += WEL * 2;
  __bf16* WoT = (__bf16*)p;                 p += WEL * 2;
  __bf16* Qp  = (__bf16*)p;                 p += ACT * 2;
  __bf16* Kp  = (__bf16*)p;                 p += ACT * 2;
  __bf16* Vp  = (__bf16*)p;                 p += ACT * 2;
  __bf16* Ao  = (__bf16*)p;                 p += ACT * 2;

  cast_f32_bf16<<<dim3(2048), 256, 0, stream>>>(q, Xq, (int)ACT);
  cast_f32_bf16<<<dim3(2048), 256, 0, stream>>>(k, Xk, (int)ACT);
  cast_f32_bf16<<<dim3(2048), 256, 0, stream>>>(v, Xv, (int)ACT);

  transpose_cast<<<dim3(64, 64), 256, 0, stream>>>(Wq, WqT);
  transpose_cast<<<dim3(64, 64), 256, 0, stream>>>(Wk, WkT);
  transpose_cast<<<dim3(64, 64), 256, 0, stream>>>(Wv, WvT);
  transpose_cast<<<dim3(64, 64), 256, 0, stream>>>(Wo, WoT);

  const dim3 gg(D_MODEL / 256, M_TOT / 128);      // (8, 32)
  gemm_bf16<true><<<gg, 256, 0, stream>>>(Xq, WqT, bq, (void*)Qp,
                                          M_TOT, D_MODEL, D_MODEL);
  gemm_bf16<true><<<gg, 256, 0, stream>>>(Xk, WkT, bk, (void*)Kp,
                                          M_TOT, D_MODEL, D_MODEL);
  gemm_bf16<true><<<gg, 256, 0, stream>>>(Xv, WvT, bv, (void*)Vp,
                                          M_TOT, D_MODEL, D_MODEL);

  attn_fused<<<dim3(SEQ / 64, BATCH * N_HEADS), 128, 0, stream>>>(Qp, Kp, Vp,
                                                                  Ao);

  gemm_bf16<false><<<gg, 256, 0, stream>>>(Ao, WoT, bo, (void*)out,
                                           M_TOT, D_MODEL, D_MODEL);
}
